// BalancedSpikingNetwork_18176301596898
// MI455X (gfx1250) — matrix operand order repaired and checked by reading, not compiled, not run on hardware
//
#include <hip/hip_runtime.h>
#include <hip/hip_fp16.h>

typedef __attribute__((ext_vector_type(16))) _Float16 v16h;
typedef __attribute__((ext_vector_type(8)))  _Float16 v8h;
typedef __attribute__((ext_vector_type(8)))  float    v8f;

// ---------------- problem constants ----------------
constexpr int B_   = 64;
constexpr int T_   = 512;
constexpr int IN_  = 128;
constexpr int NE   = 2048;
constexpr int NI   = 512;
constexpr int OUT_ = 3;

constexpr int NTOT = NE + NI;          // 2560 neurons total
constexpr int KTOT = IN_ + NTOT;       // 2688 presynaptic K (x | s_e | s_i)
constexpr int KC   = KTOT / 32;        // 84 WMMA K-chunks of 32
constexpr int KPAD = KTOT + 8;         // LDS row pitch in halves (bank-skewed)

constexpr int NGROUP = 4;              // batch groups
constexpr int GB     = 16;             // batches per group (WMMA M)
constexpr int NSLICE = 16;             // workgroups per group
constexpr int TILES  = NTOT / 16;      // 160 neuron tiles
constexpr int WAVES  = 10;             // 8 E-tiles + 2 I-tiles per WG
constexpr int BLOCK  = WAVES * 32;     // 320 threads (wave32)

constexpr float DT_         = 1.0f;
constexpr float TAU_E_INV   = 1.0f / 20.0f;
constexpr float TAU_I_INV   = 1.0f / 10.0f;
constexpr float TAU_SYN_INV = 1.0f / 5.0f;
constexpr float V_TH_       = 1.0f;
constexpr float V_RESET_    = 0.0f;
constexpr float V_LEAK_     = 0.0f;

// ---------------- workspace layout ----------------
constexpr size_t BFRAG_ELEMS = (size_t)TILES * KC * 32 * 16;      // 6,881,280 halves
constexpr size_t BFRAG_BYTES = BFRAG_ELEMS * 2;                   // ~13.76 MB
constexpr size_t SPIKE_ELEMS = (size_t)NGROUP * GB * NTOT;        // one buffer
constexpr size_t SPIKE_BYTES = SPIKE_ELEMS * 2;                   // 320 KB
constexpr size_t RATES_BYTES = (size_t)B_ * NE * 4;               // 512 KB
constexpr size_t OFF_SPIKES  = BFRAG_BYTES;                       // 256B-aligned
constexpr size_t OFF_RATES   = OFF_SPIKES + 2 * SPIKE_BYTES;
constexpr size_t OFF_CTR     = OFF_RATES + RATES_BYTES;

constexpr int SMEM_BYTES = GB * KPAD * 2;                         // 86,272 B dynamic LDS

// ---------------------------------------------------------------------------
// Pack concatenated, relu'd, sign-folded weights into exact per-lane WMMA-B
// fragment order: Bfrag[tile][kc][lane][e], where for V_WMMA_F32_16X16X32_F16
// the B (K=32 x N=16) operand has: n_local = lane&15, k_local = 8*(lane>=16)+(e&7)+16*(e>=8).
// Hot-loop load is then one contiguous 32B read per lane.
// ---------------------------------------------------------------------------
__global__ void pack_weights_kernel(const float* __restrict__ W_ee,
                                    const float* __restrict__ W_ie,
                                    const float* __restrict__ W_ei,
                                    const float* __restrict__ W_ii,
                                    const float* __restrict__ W_e_in,
                                    const float* __restrict__ W_i_in,
                                    _Float16* __restrict__ Bfrag) {
  size_t idx = (size_t)blockIdx.x * blockDim.x + threadIdx.x;
  if (idx >= BFRAG_ELEMS) return;
  int e    = (int)(idx & 15);
  int lane = (int)((idx >> 4) & 31);
  int kc   = (int)((idx >> 9) % KC);
  int tile = (int)(idx / ((size_t)KC * 512));

  int n_local = lane & 15;
  int k       = kc * 32 + ((lane >> 4) << 3) + (e & 7) + ((e >> 3) << 4);   // global K index 0..2687

  float w;
  if (tile < NE / 16) {                 // excitatory postsynaptic neuron
    int n = tile * 16 + n_local;
    if (k < IN_) {
      w = W_e_in[(size_t)n * IN_ + k];                       // raw (no relu)
    } else if (k < IN_ + NE) {
      w = W_ee[(size_t)n * NE + (k - IN_)]; w = w > 0.f ? w : 0.f;
    } else {
      w = W_ei[(size_t)n * NI + (k - IN_ - NE)]; w = w > 0.f ? -w : 0.f; // minus folded
    }
  } else {                              // inhibitory postsynaptic neuron
    int n = (tile - NE / 16) * 16 + n_local;
    if (k < IN_) {
      w = W_i_in[(size_t)n * IN_ + k];
    } else if (k < IN_ + NE) {
      w = W_ie[(size_t)n * NE + (k - IN_)]; w = w > 0.f ? w : 0.f;
    } else {
      w = W_ii[(size_t)n * NI + (k - IN_ - NE)]; w = w > 0.f ? -w : 0.f;
    }
  }
  Bfrag[idx] = (_Float16)w;
}

// ---------------------------------------------------------------------------
// Persistent SNN kernel. grid = (16 slices, 4 batch groups), 320 threads.
// One release/acquire global barrier per group per timestep; spike exchange
// is double-buffered in L2-resident global memory. Spike panel is staged into
// LDS with gfx1250 async-to-LDS DMA (ASYNCcnt) while x_t is converted f32->f16.
// ---------------------------------------------------------------------------
__global__ __launch_bounds__(BLOCK, 1)
void snn_main_kernel(const float* __restrict__ x,
                     const _Float16* __restrict__ Bfrag,
                     _Float16* __restrict__ spikes,   // 2 buffers of SPIKE_ELEMS
                     float* __restrict__ rates,       // [B, NE]
                     unsigned* __restrict__ barrier_ctr) {
  extern __shared__ _Float16 smem[];   // [GB][KPAD] activation panel (f16)

  const int slice = blockIdx.x;        // 0..15
  const int g     = blockIdx.y;        // 0..3
  const int tid   = threadIdx.x;
  const int lane  = tid & 31;
  const int wave  = tid >> 5;          // 0..9

  const int  tile   = (wave < 8) ? slice * 8 + wave
                                 : (NE / 16) + slice * 2 + (wave - 8);
  const bool isE    = (wave < 8);
  const float tau_i = isE ? TAU_E_INV : TAU_I_INV;
  const int  n_base = (tile < NE / 16) ? tile * 16
                                       : NE + (tile - NE / 16) * 16; // 0..2559

  unsigned* ctr = barrier_ctr + g;

  // pre-pack per-lane pointers
  const int hi8 = (lane & 16) ? 8 : 0;
  const _Float16* arow = &smem[(size_t)(lane & 15) * KPAD + hi8];
  const _Float16* bptr = Bfrag + (size_t)tile * (KC * 512) + lane * 16;

  // Per-thread async staging addresses: thread copies 16B chunk `tid` of each
  // of the 16 panel rows. LDS VGPR operand wants the 32-bit LDS byte offset,
  // which is the low 32 bits of the generic (flat) LDS-aperture address.
  const unsigned lds_row0 =
      (unsigned)(uintptr_t)&smem[IN_ + (size_t)tid * 8];            // row m=0

  // LIF state in WMMA C/D register layout (lane holds neuron n_base+(lane&15),
  // element r holds batch m = r + 8*(lane>=16))
  v8f v   = {};
  v8f cur = {};
  v8f cnt = {};

  for (int t = 0; t < T_; ++t) {
    // -------- inter-WG barrier: wait until group finished writing step t-1 --
    if (t > 0) {
      if (tid == 0) {
        while (__hip_atomic_load(ctr, __ATOMIC_ACQUIRE,
                                 __HIP_MEMORY_SCOPE_AGENT) < (unsigned)(NSLICE * t)) {
          __builtin_amdgcn_s_sleep(1);
        }
      }
      __syncthreads();
    }

    // -------- stage activation panel [GB][KTOT] into LDS -------------------
    // spikes (f16): async global->LDS DMA, one 16B chunk per thread per row.
    {
      const _Float16* srd = spikes + (size_t)(t & 1) * SPIKE_ELEMS
                                   + (size_t)g * GB * NTOT + (size_t)tid * 8;
#pragma unroll
      for (int m = 0; m < GB; ++m) {
        unsigned       ldst = lds_row0 + (unsigned)(m * KPAD * 2);
        const _Float16* gsrc = srd + (size_t)m * NTOT;
        asm volatile("global_load_async_to_lds_b128 %0, %1, off"
                     :: "v"(ldst), "v"(gsrc)
                     : "memory");
      }
    }
    // x_t (f32 -> f16), K = 0..127 — overlaps with the async DMA above
    for (int i = tid; i < GB * IN_; i += BLOCK) {
      int m = i >> 7, k = i & 127;
      smem[(size_t)m * KPAD + k] =
          (_Float16)x[((size_t)(g * GB + m) * T_ + t) * IN_ + k];
    }
    asm volatile("s_wait_asynccnt 0" ::: "memory");
    __syncthreads();

    // -------- 16x16 output tile: 84 chained v_wmma_f32_16x16x32_f16 --------
    v8f acc0 = {}, acc1 = {};
#pragma unroll 4
    for (int kc = 0; kc < KC; ++kc) {
      // A fragment (M=16 batches x K=32): two contiguous 16B LDS reads/lane
      v8h alo = *(const v8h*)(arow + kc * 32);
      v8h ahi = *(const v8h*)(arow + kc * 32 + 16);
      v16h a = __builtin_shufflevector(alo, ahi, 0, 1, 2, 3, 4, 5, 6, 7,
                                       8, 9, 10, 11, 12, 13, 14, 15);
      // B fragment: pre-packed, one 32B contiguous read/lane
      v16h b = *(const v16h*)(bptr + (size_t)kc * 512);
      // WGP-scope prefetch (default scope 0): pulls upcoming B fragments from
      // GL2 into the WGP cache — the only scope that stages closer than L2.
      {
        const _Float16* pf = bptr + (size_t)(kc + 6) * 512;
        asm volatile("global_prefetch_b8 %0, off" :: "v"(pf));
      }
      if (kc & 1)
        acc1 = __builtin_amdgcn_wmma_f32_16x16x32_f16(false, a, false, b,
                                                      (short)0, acc1, false, false);
      else
        acc0 = __builtin_amdgcn_wmma_f32_16x16x32_f16(false, a, false, b,
                                                      (short)0, acc0, false, false);
    }

    // -------- LIF update + spike emission ----------------------------------
    _Float16* swr = spikes + (size_t)((t + 1) & 1) * SPIKE_ELEMS
                           + (size_t)g * GB * NTOT;
    const int ncol = n_base + (lane & 15);
#pragma unroll
    for (int r = 0; r < 8; ++r) {
      float I_in  = acc0[r] + acc1[r];
      float v_dec = v[r] + DT_ * tau_i * ((V_LEAK_ - v[r]) + cur[r]);
      float i_dec = cur[r] * (1.0f - DT_ * TAU_SYN_INV);
      float z     = (v_dec > V_TH_) ? 1.0f : 0.0f;
      v[r]   = (z > 0.0f) ? V_RESET_ : v_dec;
      cur[r] = i_dec + I_in;
      if (isE) cnt[r] += z;
      int mrow = r + ((lane & 16) >> 1);            // batch-in-group 0..15
      swr[(size_t)mrow * NTOT + ncol] = (_Float16)z;
    }

    // -------- arrive: release our step-t spike writes ----------------------
    __syncthreads();
    if (tid == 0) {
      __hip_atomic_fetch_add(ctr, 1u, __ATOMIC_RELEASE, __HIP_MEMORY_SCOPE_AGENT);
    }
  }

  // -------- emit firing rates for readout ----------------------------------
  if (isE) {
#pragma unroll
    for (int r = 0; r < 8; ++r) {
      int mrow = r + ((lane & 16) >> 1);
      rates[(size_t)(g * GB + mrow) * NE + n_base + (lane & 15)] =
          cnt[r] * (1.0f / (float)T_);
    }
  }
}

// ---------------------------------------------------------------------------
// out[b][o] = rates[b] . readout_w[o] + readout_b[o]
// ---------------------------------------------------------------------------
__global__ void readout_kernel(const float* __restrict__ rates,
                               const float* __restrict__ rw,
                               const float* __restrict__ rb,
                               float* __restrict__ out) {
  __shared__ float red[3 * 256];
  const int b = blockIdx.x, tid = threadIdx.x;
  float p0 = 0.f, p1 = 0.f, p2 = 0.f;
  for (int n = tid; n < NE; n += 256) {
    float r = rates[(size_t)b * NE + n];
    p0 += r * rw[n];
    p1 += r * rw[NE + n];
    p2 += r * rw[2 * NE + n];
  }
  red[tid] = p0; red[256 + tid] = p1; red[512 + tid] = p2;
  __syncthreads();
  for (int s = 128; s > 0; s >>= 1) {
    if (tid < s) {
      red[tid]       += red[tid + s];
      red[256 + tid] += red[256 + tid + s];
      red[512 + tid] += red[512 + tid + s];
    }
    __syncthreads();
  }
  if (tid < OUT_) out[b * OUT_ + tid] = red[tid * 256] + rb[tid];
}

// ---------------------------------------------------------------------------
extern "C" void kernel_launch(void* const* d_in, const int* in_sizes, int n_in,
                              void* d_out, int out_size, void* d_ws, size_t ws_size,
                              hipStream_t stream) {
  (void)in_sizes; (void)n_in; (void)out_size; (void)ws_size;
  const float* x      = (const float*)d_in[0];
  const float* W_ee   = (const float*)d_in[1];
  const float* W_ie   = (const float*)d_in[2];
  const float* W_ei   = (const float*)d_in[3];
  const float* W_ii   = (const float*)d_in[4];
  const float* W_e_in = (const float*)d_in[5];
  const float* W_i_in = (const float*)d_in[6];
  const float* rw     = (const float*)d_in[7];
  const float* rb     = (const float*)d_in[8];
  float* out = (float*)d_out;

  char* ws = (char*)d_ws;
  _Float16* Bfrag  = (_Float16*)ws;
  _Float16* spikes = (_Float16*)(ws + OFF_SPIKES);
  float*    rates  = (float*)(ws + OFF_RATES);
  unsigned* ctr    = (unsigned*)(ws + OFF_CTR);

  // step-0 spike buffer must be zero; counters must restart each launch
  hipMemsetAsync(spikes, 0, SPIKE_BYTES, stream);
  hipMemsetAsync(ctr, 0, NGROUP * sizeof(unsigned), stream);

  {
    int threads = 256;
    int blocks  = (int)((BFRAG_ELEMS + threads - 1) / threads);
    pack_weights_kernel<<<blocks, threads, 0, stream>>>(W_ee, W_ie, W_ei, W_ii,
                                                        W_e_in, W_i_in, Bfrag);
  }

  hipFuncSetAttribute(reinterpret_cast<const void*>(snn_main_kernel),
                      hipFuncAttributeMaxDynamicSharedMemorySize, SMEM_BYTES);
  snn_main_kernel<<<dim3(NSLICE, NGROUP), BLOCK, SMEM_BYTES, stream>>>(
      x, Bfrag, spikes, rates, ctr);

  readout_kernel<<<B_, 256, 0, stream>>>(rates, rw, rb, out);
}
